// Generator_70935679861232
// MI455X (gfx1250) — matrix operand order repaired and checked by reading, not compiled, not run on hardware
//
#include <hip/hip_runtime.h>
#include <hip/hip_bf16.h>
#include <math.h>

// Problem constants (match reference)
#define B_    8
#define N_    256
#define H_    512
#define E_    512
#define RD_   64
#define V_    32000
#define KX_   (RD_ + E_)   // 576
#define G4_   (4 * H_)     // 2048
#define ROWS_ (B_ * N_)    // 2048

typedef __bf16 bf16_t;
typedef __bf16 v16bf __attribute__((ext_vector_type(16)));
typedef __bf16 v8bf  __attribute__((ext_vector_type(8)));
typedef float  v8f   __attribute__((ext_vector_type(8)));

// ---------------------------------------------------------------------------
// WMMA fragment loaders (bf16, 16x16x32, wave32).
// A-matrix (16xK): lane L holds row M = L&15; for K-chunk kc,
//   its 16 elems are contiguous at kc*32 + (L>=16 ? 16 : 0)
//   (two 8-elem loads, matching B's k order).
// B-matrix (Kx16): lane L holds col N = L&15; 16 contiguous elems at
//   kc*32 + (L>=16 ? 16 : 0).
// With (col x K) transposed weights, both are plain vector loads.
// ---------------------------------------------------------------------------
__device__ inline v16bf load_a_frag(const bf16_t* rowp, int lane) {
  const int off = (lane & 16) ? 16 : 0;
  v8bf lo = *(const v8bf*)(rowp + off);
  v8bf hi = *(const v8bf*)(rowp + off + 8);
  v16bf r;
#pragma unroll
  for (int j = 0; j < 8; ++j) { r[j] = lo[j]; r[j + 8] = hi[j]; }
  return r;
}

__device__ inline v16bf load_b_frag(const bf16_t* colp, int lane) {
  const int off = (lane & 16) ? 16 : 0;
  return *(const v16bf*)(colp + off);
}

// ---------------------------------------------------------------------------
// Weight transpose + f32 -> bf16 convert: dst[n*K + k] = src[k*Nc + n]
// ---------------------------------------------------------------------------
__global__ void k_conv_transpose(const float* __restrict__ src,
                                 bf16_t* __restrict__ dst,
                                 int K, int Nc, long total) {
  long i = (long)blockIdx.x * blockDim.x + threadIdx.x;
  if (i >= total) return;
  int k = (int)(i % K);
  long n = i / K;
  dst[i] = (bf16_t)src[(long)k * Nc + n];
}

// ---------------------------------------------------------------------------
// Gather x = concat(rel_emb[rel_ids], word_emb[pword]) -> bf16, rows r = n*8+b
// ---------------------------------------------------------------------------
__global__ void k_gather_x(const int* __restrict__ rel_ids,
                           const int* __restrict__ sent_ids,
                           const int* __restrict__ parent,
                           const float* __restrict__ word_emb,
                           const float* __restrict__ rel_emb,
                           bf16_t* __restrict__ xbf, long total) {
  long i = (long)blockIdx.x * blockDim.x + threadIdx.x;
  if (i >= total) return;
  int k = (int)(i % KX_);
  int r = (int)(i / KX_);
  int n = r >> 3, b = r & 7;
  float v;
  if (k < RD_) {
    v = rel_emb[(long)rel_ids[b * N_ + n] * RD_ + k];
  } else {
    int p = parent[n];
    int w = (p < 0) ? 0 : sent_ids[b * N_ + p];   // SOS == 0
    v = word_emb[(long)w * E_ + (k - RD_)];
  }
  xbf[i] = (bf16_t)v;
}

// Hs/Cs slot 0 = root_h (reference sets Cs0 = Hs0)
__global__ void k_init_state(const float* __restrict__ root_h,
                             float* __restrict__ Hs, float* __restrict__ Cs) {
  int i = blockIdx.x * blockDim.x + threadIdx.x;
  if (i >= B_ * H_) return;
  int b = i >> 9, k = i & 511;
  long idx = ((long)b * (N_ + 1)) * H_ + k;  // slot 0
  Hs[idx] = root_h[i];
  Cs[idx] = root_h[i];
}

// ---------------------------------------------------------------------------
// XgT[r][g] = (x @ Wx)[r][g] + b[g], r = n*8+b.  4 waves/block, 16x64 tile.
// ---------------------------------------------------------------------------
__global__ __launch_bounds__(128) void k_gemm_xg(
    const bf16_t* __restrict__ xbf, const bf16_t* __restrict__ wxT,
    const float* __restrict__ bias, float* __restrict__ xgT) {
  const int lane  = threadIdx.x & 31;
  const int wave  = threadIdx.x >> 5;
  const int mtile = blockIdx.x;
  const int col0  = blockIdx.y * 64 + wave * 16;
  const int m     = lane & 15;
  const bf16_t* arow = xbf + (long)(mtile * 16 + m) * KX_;
  const bf16_t* bcol = wxT + (long)(col0 + m) * KX_;
  v8f acc = {};
#pragma unroll
  for (int kc = 0; kc < KX_ / 32; ++kc) {
    if (kc + 1 < KX_ / 32) __builtin_prefetch(bcol + (kc + 1) * 32, 0, 1);
    v16bf a  = load_a_frag(arow + kc * 32, lane);
    v16bf bm = load_b_frag(bcol + kc * 32, lane);
    acc = __builtin_amdgcn_wmma_f32_16x16x32_bf16(false, a, false, bm,
                                                  (short)0, acc, false, false);
  }
  const int row0 = mtile * 16 + ((lane & 16) ? 8 : 0);
  const int col  = col0 + m;
  const float bc = bias[col];
#pragma unroll
  for (int v = 0; v < 8; ++v)
    xgT[(long)(row0 + v) * G4_ + col] = acc[v] + bc;
}

// ---------------------------------------------------------------------------
// Persistent tree-LSTM scan: one workgroup, 32 waves. Each step:
//   gather parent h/c -> LDS (bf16 A-tile, rows 8..15 zero), then each wave
//   computes 16 hidden columns of ALL FOUR gates (4 WMMA chains), fuses the
//   LSTM nonlinearity in registers, and stores h/c for children.
// ---------------------------------------------------------------------------
__global__ __launch_bounds__(1024) void k_scan(
    const int* __restrict__ parent, const bf16_t* __restrict__ whT,
    const float* __restrict__ xgT, float* __restrict__ Hs,
    float* __restrict__ Cs, bf16_t* __restrict__ hsbf) {
  __shared__ bf16_t Abuf[16][H_];   // 16 KB: padded parent-h tile (bf16)
  __shared__ float  Pc[B_][H_];     // 16 KB: parent c (f32)

  const int t    = threadIdx.x;
  const int lane = t & 31;
  const int wave = t >> 5;          // 0..31 -> owns 16 hidden columns
  const int m    = lane & 15;

  // rows 8..15 of the A tile are the zero padding (B=8 < 16)
  for (int i = t; i < 8 * H_; i += 1024) Abuf[8 + (i >> 9)][i & 511] = (bf16_t)0.f;

  for (int step = 0; step < N_; ++step) {
    const int slot = parent[step] + 1;
    __syncthreads();   // previous step's compute done before refilling LDS
    for (int i = t; i < B_ * H_; i += 1024) {
      int b = i >> 9, k = i & 511;
      long sidx = ((long)b * (N_ + 1) + slot) * H_ + k;
      Abuf[b][k] = (bf16_t)Hs[sidx];
      Pc[b][k]   = Cs[sidx];
    }
    __syncthreads();

    const int j0 = wave * 16;
    v8f aI = {}, aF = {}, aO = {}, aG = {};
    const bf16_t* bI = whT + (long)(0 * H_ + j0 + m) * H_;
    const bf16_t* bF = whT + (long)(1 * H_ + j0 + m) * H_;
    const bf16_t* bO = whT + (long)(2 * H_ + j0 + m) * H_;
    const bf16_t* bG = whT + (long)(3 * H_ + j0 + m) * H_;
#pragma unroll
    for (int kc = 0; kc < H_ / 32; ++kc) {
      v16bf a = load_a_frag(&Abuf[m][kc * 32], lane);
      aI = __builtin_amdgcn_wmma_f32_16x16x32_bf16(false, a, false,
            load_b_frag(bI + kc * 32, lane), (short)0, aI, false, false);
      aF = __builtin_amdgcn_wmma_f32_16x16x32_bf16(false, a, false,
            load_b_frag(bF + kc * 32, lane), (short)0, aF, false, false);
      aO = __builtin_amdgcn_wmma_f32_16x16x32_bf16(false, a, false,
            load_b_frag(bO + kc * 32, lane), (short)0, aO, false, false);
      aG = __builtin_amdgcn_wmma_f32_16x16x32_bf16(false, a, false,
            load_b_frag(bG + kc * 32, lane), (short)0, aG, false, false);
    }

    if (lane < 16) {   // rows 0..7 (the real batches) live in lanes 0..15
      const int col = j0 + lane;
#pragma unroll
      for (int v = 0; v < B_; ++v) {
        const long xrow = (long)(step * B_ + v) * G4_;
        float gi = aI[v] + xgT[xrow + 0 * H_ + col];
        float gf = aF[v] + xgT[xrow + 1 * H_ + col];
        float go = aO[v] + xgT[xrow + 2 * H_ + col];
        float gg = aG[v] + xgT[xrow + 3 * H_ + col];
        float si = 1.f / (1.f + __expf(-gi));
        float sf = 1.f / (1.f + __expf(-gf));
        float so = 1.f / (1.f + __expf(-go));
        float c  = sf * Pc[v][col] + si * tanhf(gg);
        float h  = so * tanhf(c);
        long didx = ((long)v * (N_ + 1) + step + 1) * H_ + col;
        Hs[didx] = h;
        Cs[didx] = c;
        hsbf[((long)v * N_ + step) * H_ + col] = (bf16_t)h;   // row = b*256+n
      }
    }
    __threadfence();   // make Hs/Cs stores visible before next gather
  }
}

// ---------------------------------------------------------------------------
// logits = hs @ Wo + bo : (2048 x 512) @ (512 x 32000). 8 waves/block,
// 16x128 tile. Memory-bound (262 MB f32 output); bf16 WMMA compute.
// ---------------------------------------------------------------------------
__global__ __launch_bounds__(256) void k_gemm_out(
    const bf16_t* __restrict__ hsbf, const bf16_t* __restrict__ woT,
    const float* __restrict__ bo, float* __restrict__ out) {
  const int lane  = threadIdx.x & 31;
  const int wave  = threadIdx.x >> 5;
  const int mtile = blockIdx.x;
  const int col0  = blockIdx.y * 128 + wave * 16;
  const int m     = lane & 15;
  const bf16_t* arow = hsbf + (long)(mtile * 16 + m) * H_;
  const bf16_t* bcol = woT + (long)(col0 + m) * H_;
  v8f acc = {};
#pragma unroll
  for (int kc = 0; kc < H_ / 32; ++kc) {
    if (kc + 1 < H_ / 32) __builtin_prefetch(bcol + (kc + 1) * 32, 0, 1);
    v16bf a  = load_a_frag(arow + kc * 32, lane);
    v16bf bm = load_b_frag(bcol + kc * 32, lane);
    acc = __builtin_amdgcn_wmma_f32_16x16x32_bf16(false, a, false, bm,
                                                  (short)0, acc, false, false);
  }
  const int row0 = mtile * 16 + ((lane & 16) ? 8 : 0);  // row = b*256+n
  const int col  = col0 + m;
  const float bc = bo[col];
#pragma unroll
  for (int v = 0; v < 8; ++v)
    out[(long)(row0 + v) * V_ + col] = acc[v] + bc;      // out[b][n][v]
}

// ---------------------------------------------------------------------------
extern "C" void kernel_launch(void* const* d_in, const int* in_sizes, int n_in,
                              void* d_out, int out_size, void* d_ws,
                              size_t ws_size, hipStream_t stream) {
  const int*   rel_ids  = (const int*)d_in[0];
  const int*   sent_ids = (const int*)d_in[1];
  const int*   parent   = (const int*)d_in[2];
  const float* root_h   = (const float*)d_in[3];
  const float* word_emb = (const float*)d_in[4];
  const float* rel_emb  = (const float*)d_in[5];
  const float* Wx       = (const float*)d_in[6];
  const float* Wh       = (const float*)d_in[7];
  const float* bias     = (const float*)d_in[8];
  const float* Wo       = (const float*)d_in[9];
  const float* bo       = (const float*)d_in[10];
  float* out = (float*)d_out;
  (void)in_sizes; (void)n_in; (void)out_size; (void)ws_size;

  char* ws = (char*)d_ws;
  size_t cur = 0;
  auto wsalloc = [&](size_t bytes) -> void* {
    void* p = ws + cur;
    cur = (cur + bytes + 255) & ~(size_t)255;
    return p;
  };
  bf16_t* wxT  = (bf16_t*)wsalloc((size_t)G4_ * KX_ * 2);        // 2.25 MB
  bf16_t* whT  = (bf16_t*)wsalloc((size_t)G4_ * H_ * 2);         // 2 MB
  bf16_t* woT  = (bf16_t*)wsalloc((size_t)V_ * H_ * 2);          // 32.8 MB
  bf16_t* xbf  = (bf16_t*)wsalloc((size_t)ROWS_ * KX_ * 2);      // 2.25 MB
  float*  xgT  = (float*)wsalloc((size_t)ROWS_ * G4_ * 4);       // 16.8 MB
  float*  Hs   = (float*)wsalloc((size_t)B_ * (N_ + 1) * H_ * 4);// 4.2 MB
  float*  Cs   = (float*)wsalloc((size_t)B_ * (N_ + 1) * H_ * 4);// 4.2 MB
  bf16_t* hsbf = (bf16_t*)wsalloc((size_t)ROWS_ * H_ * 2);       // 2.1 MB

  { long tot = (long)G4_ * KX_;
    k_conv_transpose<<<(tot + 255) / 256, 256, 0, stream>>>(Wx, wxT, KX_, G4_, tot); }
  { long tot = (long)G4_ * H_;
    k_conv_transpose<<<(tot + 255) / 256, 256, 0, stream>>>(Wh, whT, H_, G4_, tot); }
  { long tot = (long)V_ * H_;
    k_conv_transpose<<<(tot + 255) / 256, 256, 0, stream>>>(Wo, woT, H_, V_, tot); }
  { long tot = (long)ROWS_ * KX_;
    k_gather_x<<<(tot + 255) / 256, 256, 0, stream>>>(rel_ids, sent_ids, parent,
                                                      word_emb, rel_emb, xbf, tot); }
  k_init_state<<<(B_ * H_ + 255) / 256, 256, 0, stream>>>(root_h, Hs, Cs);

  dim3 g1(ROWS_ / 16, G4_ / 64);
  k_gemm_xg<<<g1, 128, 0, stream>>>(xbf, wxT, bias, xgT);

  k_scan<<<1, 1024, 0, stream>>>(parent, whT, xgT, Hs, Cs, hsbf);

  dim3 g2(ROWS_ / 16, V_ / 128);
  k_gemm_out<<<g2, 256, 0, stream>>>(hsbf, woT, bo, out);
}
